// MultiSampler_14620068675686
// MI455X (gfx1250) — matrix operand order repaired and checked by reading, not compiled, not run on hardware
//
#include <hip/hip_runtime.h>
#include <hip/hip_bf16.h>

#define DEVINL __device__ __forceinline__

// Fixed problem shape from setup_inputs()
constexpr int BB   = 2;
constexpr int Himg = 512, Wimg = 512;
constexpr int Npt  = 2500;
constexpr int Mpt  = 4096;
constexpr int C    = 128;
constexpr int H4   = Himg / 4, W4 = Wimg / 4;
constexpr int BN   = BB * Npt;           // 5000
constexpr int P    = 29;                  // |{(i,j): i^2+j^2 <= 9}|
constexpr int Q    = 80;                  // |{(i,j): 25 <= i^2+j^2 <= 49}|
constexpr int SC   = 1 + Q + BN;          // 5081 columns per score row
constexpr float NEG_D2 = 25.0f;

// ---- offset tables (exact reference order: j outer, i inner; dy=i, dx=j) ----
struct OffTab { float py[P], px[P], ny[Q], nx[Q]; };
constexpr OffTab make_offsets() {
  OffTab t{}; int np = 0, nq = 0;
  for (int j = -49; j <= 49; ++j)
    for (int i = -49; i <= 49; ++i) {
      int d2 = i * i + j * j;
      if (d2 <= 9)                  { t.py[np] = (float)i; t.px[np] = (float)j; ++np; }
      else if (d2 >= 25 && d2 <= 49){ t.ny[nq] = (float)i; t.nx[nq] = (float)j; ++nq; }
    }
  return t;
}
constexpr OffTab h_off = make_offsets();
__constant__ OffTab g_off = h_off;

// ---- vector types ----
typedef _Float16 v16h __attribute__((ext_vector_type(16)));
typedef _Float16 v8h  __attribute__((ext_vector_type(8)));
typedef float    v8f  __attribute__((ext_vector_type(8)));
typedef float    v4f  __attribute__((ext_vector_type(4)));

DEVINL float clampf(float v, float lo, float hi) { return fminf(fmaxf(v, lo), hi); }
DEVINL float dot4(v4f a, v4f b) { return a[0]*b[0] + a[1]*b[1] + a[2]*b[2] + a[3]*b[3]; }

DEVINL float wsum(float v) {
#pragma unroll
  for (int m = 1; m < 32; m <<= 1) v += __shfl_xor(v, m, 32);
  return v;
}
DEVINL void wsum2(float& a, float& b) {
#pragma unroll
  for (int m = 1; m < 32; m <<= 1) { a += __shfl_xor(a, m, 32); b += __shfl_xor(b, m, 32); }
}
DEVINL void wsum3(float& a, float& b, float& c) {
#pragma unroll
  for (int m = 1; m < 32; m <<= 1) {
    a += __shfl_xor(a, m, 32); b += __shfl_xor(b, m, 32); c += __shfl_xor(c, m, 32);
  }
}

// bilinear sample of 4 consecutive channels (channel-last layout), coords are
// full-res (divided by 4 here), floor clipped exactly like the reference
DEVINL v4f sampleFeat(const float* img, float y, float x, int cb) {
  float u = y * 0.25f, v = x * 0.25f;
  int i0 = (int)floorf(u); i0 = i0 < 0 ? 0 : (i0 > H4 - 1 ? H4 - 1 : i0);
  int j0 = (int)floorf(v); j0 = j0 < 0 ? 0 : (j0 > W4 - 1 ? W4 - 1 : j0);
  int i1 = i0 + 1 > H4 - 1 ? H4 - 1 : i0 + 1;
  int j1 = j0 + 1 > W4 - 1 ? W4 - 1 : j0 + 1;
  float di = u - (float)i0, dj = v - (float)j0;
  v4f a = *(const v4f*)(img + ((size_t)i0 * W4 + j0) * C + cb);
  v4f b = *(const v4f*)(img + ((size_t)i0 * W4 + j1) * C + cb);
  v4f c = *(const v4f*)(img + ((size_t)i1 * W4 + j0) * C + cb);
  v4f d = *(const v4f*)(img + ((size_t)i1 * W4 + j1) * C + cb);
  return a * ((1.f - di) * (1.f - dj)) + b * ((1.f - di) * dj) +
         c * (di * (1.f - dj))         + d * (di * dj);
}

DEVINL float sampleConf(const float* cmap, float y, float x) {
  float u = y * 0.25f, v = x * 0.25f;
  int i0 = (int)floorf(u); i0 = i0 < 0 ? 0 : (i0 > H4 - 1 ? H4 - 1 : i0);
  int j0 = (int)floorf(v); j0 = j0 < 0 ? 0 : (j0 > W4 - 1 ? W4 - 1 : j0);
  int i1 = i0 + 1 > H4 - 1 ? H4 - 1 : i0 + 1;
  int j1 = j0 + 1 > W4 - 1 ? W4 - 1 : j0 + 1;
  float di = u - (float)i0, dj = v - (float)j0;
  float a = cmap[i0 * W4 + j0], b = cmap[i0 * W4 + j1];
  float c = cmap[i1 * W4 + j0], d = cmap[i1 * W4 + j1];
  return (1.f - di) * (1.f - dj) * a + (1.f - di) * dj * b + di * (1.f - dj) * c + di * dj * d;
}

// ---------------------------------------------------------------------------
// Phase 1: one wave per point (lane handles 4 channels). blockIdx.y==0: query
// points (vf0/vnf0, pos/neg ring scores, conf); blockIdx.y==1: dtr/ndtr points.
// ---------------------------------------------------------------------------
__global__ void __launch_bounds__(256)
sampler_kernel(const float* feat0, const float* feat1,
               const float* nfeat0, const float* nfeat1,
               const float* conf0, const float* conf1,
               const float* nconf0, const float* nconf1,
               const float* pos0, const float* pos1,
               const int* idx_q, const int* idx_d,
               float* out_scores, float* out_noise, float* out_mask,
               float* out_qconf, float* out_nqconf,
               _Float16* Ah, _Float16* Bh, _Float16* nAh, _Float16* nBh,
               float* wvp1x, float* wvp1y, float* wvp2x, float* wvp2y)
{
  const int lane = threadIdx.x & 31;
  const int p = blockIdx.x * 8 + (threadIdx.x >> 5);
  if (p >= BN) return;
  const int b = p / Npt, n = p - b * Npt;
  const int cb = lane * 4;
  const size_t fstride = (size_t)H4 * W4 * C;
  const size_t cstride = (size_t)H4 * W4;

  if (blockIdx.y == 1) {
    // dtr / ndtr at pos1[idx_d]
    const int idd = idx_d[b * Npt + n];
    const float y = pos1[((size_t)b * Mpt + idd) * 2 + 0];
    const float x = pos1[((size_t)b * Mpt + idd) * 2 + 1];
    if (lane == 0) { wvp2y[p] = y; wvp2x[p] = x; }
    v4f f = sampleFeat(feat1 + b * fstride, y, x, cb);
    v4f g = sampleFeat(nfeat1 + b * fstride, y, x, cb);
    float s0 = dot4(f, f), s1 = dot4(g, g);
    wsum2(s0, s1);
    float r0 = 1.f / fmaxf(sqrtf(s0), 1e-12f);
    float r1 = 1.f / fmaxf(sqrtf(s1), 1e-12f);
    _Float16* d0 = Bh  + (size_t)p * C + cb;
    _Float16* d1 = nBh + (size_t)p * C + cb;
#pragma unroll
    for (int k = 0; k < 4; ++k) { d0[k] = (_Float16)(f[k] * r0); d1[k] = (_Float16)(g[k] * r1); }
    return;
  }

  const int iq = idx_q[b * Npt + n];
  const float p0y = pos0[((size_t)b * Mpt + iq) * 2 + 0];
  const float p0x = pos0[((size_t)b * Mpt + iq) * 2 + 1];
  const float p1y = pos1[((size_t)b * Mpt + iq) * 2 + 0];
  const float p1x = pos1[((size_t)b * Mpt + iq) * 2 + 1];

  const float* f0b  = feat0  + b * fstride;
  const float* f1b  = feat1  + b * fstride;
  const float* nf0b = nfeat0 + b * fstride;
  const float* nf1b = nfeat1 + b * fstride;

  if (lane == 0) {
    wvp1y[p] = p1y; wvp1x[p] = p1x;
    bool m = (p1x >= 0.f) && (p1x < (float)Wimg) && (p1y >= 0.f) && (p1y < (float)Himg);
    out_mask[p] = m ? 1.f : 0.f;
  }

  // normalized query vectors
  v4f a0 = sampleFeat(f0b,  p0y, p0x, cb);
  v4f a1 = sampleFeat(nf0b, p0y, p0x, cb);
  float s0 = dot4(a0, a0), s1 = dot4(a1, a1);
  wsum2(s0, s1);
  v4f vf0  = a0 * (1.f / fmaxf(sqrtf(s0), 1e-12f));
  v4f vnf0 = a1 * (1.f / fmaxf(sqrtf(s1), 1e-12f));
  {
    _Float16* d0 = Ah  + (size_t)p * C + cb;
    _Float16* d1 = nAh + (size_t)p * C + cb;
#pragma unroll
    for (int k = 0; k < 4; ++k) { d0[k] = (_Float16)vf0[k]; d1[k] = (_Float16)vnf0[k]; }
  }

  // positive ring: ps and nps both use feat1 samples (reference shares f1p)
  float best1 = -1e30f, best2 = -1e30f; int bi1 = 0, bi2 = 0;
#pragma unroll 1
  for (int k = 0; k < P; ++k) {
    float sy = clampf(p1y + g_off.py[k], 0.f, (float)(Himg - 1));
    float sx = clampf(p1x + g_off.px[k], 0.f, (float)(Wimg - 1));
    v4f f = sampleFeat(f1b, sy, sx, cb);
    float ss = dot4(f, f), d1 = dot4(vf0, f), d2 = dot4(vnf0, f);
    wsum3(ss, d1, d2);
    float inv = 1.f / fmaxf(sqrtf(ss), 1e-12f);
    float sc1 = d1 * inv, sc2 = d2 * inv;
    if (sc1 > best1) { best1 = sc1; bi1 = k; }
    if (sc2 > best2) { best2 = sc2; bi2 = k; }
  }

  // negative ring: nscores (feat1 vs vf0), nnscores (noise_feat1 vs vnf0)
  // streaming, write-once -> non-temporal stores (keep 192MB L2 for gathers)
#pragma unroll 1
  for (int k = 0; k < Q; ++k) {
    float sy = clampf(p1y + g_off.ny[k], 0.f, (float)(Himg - 1));
    float sx = clampf(p1x + g_off.nx[k], 0.f, (float)(Wimg - 1));
    v4f f = sampleFeat(f1b,  sy, sx, cb);
    v4f g = sampleFeat(nf1b, sy, sx, cb);
    float ss = dot4(f, f), d1 = dot4(vf0, f), tt = dot4(g, g);
    float d2 = dot4(vnf0, g);
    wsum3(ss, d1, tt);
    d2 = wsum(d2);
    if (lane == 0) {
      __builtin_nontemporal_store(d1 / fmaxf(sqrtf(ss), 1e-12f),
                                  &out_scores[(size_t)p * SC + 1 + k]);
      __builtin_nontemporal_store(d2 / fmaxf(sqrtf(tt), 1e-12f),
                                  &out_noise [(size_t)p * SC + 1 + k]);
    }
  }

  if (lane == 0) {
    out_scores[(size_t)p * SC] = best1;
    out_noise [(size_t)p * SC] = best2;
    const float* c0b  = conf0  + b * cstride;
    const float* c1b  = conf1  + b * cstride;
    const float* nc0b = nconf0 + b * cstride;
    const float* nc1b = nconf1 + b * cstride;
    float qc  = sampleConf(c0b,  p0y, p0x);
    float nqc = sampleConf(nc0b, p0y, p0x);
    float sy = clampf(p1y + g_off.py[bi1], 0.f, (float)(Himg - 1));
    float sx = clampf(p1x + g_off.px[bi1], 0.f, (float)(Wimg - 1));
    out_qconf[p] = 0.5f * (qc + sampleConf(c1b, sy, sx));
    float ty = clampf(p1y + g_off.py[bi2], 0.f, (float)(Himg - 1));
    float tx = clampf(p1x + g_off.px[bi2], 0.f, (float)(Wimg - 1));
    out_nqconf[p] = 0.5f * (nqc + sampleConf(nc1b, ty, tx));
  }
}

// ---------------------------------------------------------------------------
// Phase 2: dscores GEMM (5000x5000, K=128) with distance masking, f16 WMMA.
// 64x64 block tile, 8 waves (4x2), each wave: 16 rows x 32 cols (2 accums).
// blockIdx.z selects normal vs noise operand/output sets.
// ---------------------------------------------------------------------------
__global__ void __launch_bounds__(256)
gemm_kernel(const _Float16* Ah, const _Float16* Bh,
            const _Float16* nAh, const _Float16* nBh,
            const float* wvp1x, const float* wvp1y,
            const float* wvp2x, const float* wvp2y,
            float* out)
{
  const _Float16* A  = blockIdx.z ? nAh : Ah;
  const _Float16* Bm = blockIdx.z ? nBh : Bh;
  float* O = out + (size_t)blockIdx.z * (size_t)BN * SC;

  __shared__ float s_r[2][64];  // vp1x, vp1y for the 64 tile rows
  __shared__ float s_c[2][64];  // vp2x, vp2y for the 64 tile cols

  const int tid = threadIdx.x;
  const int tileM = blockIdx.y * 64, tileN = blockIdx.x * 64;
  if (tid < 64) {
    int r = tileM + tid; bool ok = r < BN;
    s_r[0][tid] = ok ? wvp1x[r] : 0.f;
    s_r[1][tid] = ok ? wvp1y[r] : 0.f;
  } else if (tid < 128) {
    int c = tileN + (tid - 64); bool ok = c < BN;
    s_c[0][tid - 64] = ok ? wvp2x[c] : 0.f;
    s_c[1][tid - 64] = ok ? wvp2y[c] : 0.f;
  }
  __syncthreads();

  const int wave = tid >> 5, lane = tid & 31;
  const int wm = wave & 3, wn = wave >> 2;
  const int rowBase = tileM + wm * 16;
  const int colBase = tileN + wn * 32;
  const int lr = lane & 15, lh = lane >> 4;

  // A lane feeds row lr (both half-waves), K chunk lh*16..lh*16+15
  int arow = rowBase + lr; if (arow >= BN) arow = BN - 1;
  int c0 = colBase + lr;      int c0c = c0 >= BN ? BN - 1 : c0;
  int c1 = colBase + 16 + lr; int c1c = c1 >= BN ? BN - 1 : c1;
  const _Float16* aPtr  = A  + (size_t)arow * C;
  const _Float16* b0Ptr = Bm + (size_t)c0c * C;
  const _Float16* b1Ptr = Bm + (size_t)c1c * C;

  v8f acc0 = {}; v8f acc1 = {};
#pragma unroll
  for (int kk = 0; kk < C; kk += 32) {
    v16h a = *(const v16h*)(aPtr + kk + lh * 16);
    v16h b0 = *(const v16h*)(b0Ptr + kk + lh * 16);
    v16h b1 = *(const v16h*)(b1Ptr + kk + lh * 16);
    acc0 = __builtin_amdgcn_wmma_f32_16x16x32_f16(false, a, false, b0, (short)0, acc0, false, false);
    acc1 = __builtin_amdgcn_wmma_f32_16x16x32_f16(false, a, false, b1, (short)0, acc1, false, false);
  }

  // store with distance mask; D layout: element (vgpr r, lane) -> M = lh*8+r, N = lr
  // streaming 200MB write-once region -> non-temporal stores
#pragma unroll
  for (int t = 0; t < 2; ++t) {
    int col = colBase + t * 16 + lr;
    if (col >= BN) continue;
    float cx = s_c[0][col - tileN], cy = s_c[1][col - tileN];
    int bcol = col / Npt;
#pragma unroll
    for (int r = 0; r < 8; ++r) {
      int row = rowBase + lh * 8 + r;
      if (row >= BN) continue;
      float rx = s_r[0][row - tileM], ry = s_r[1][row - tileM];
      float dx = cx - rx, dy = cy - ry;
      float d2 = dx * dx + dy * dy + ((row / Npt) != bcol ? NEG_D2 : 0.f);
      float v = t ? acc1[r] : acc0[r];
      if (d2 < NEG_D2) v = 0.f;
      __builtin_nontemporal_store(v, &O[(size_t)row * SC + 1 + Q + col]);
    }
  }
}

// gt: one-hot first column, zeros elsewhere (HBM-store-bound fill, NT stores)
__global__ void gt_kernel(float* gt) {
  size_t row = blockIdx.x;
  float* g = gt + row * (size_t)SC;
  for (int c = threadIdx.x; c < SC; c += blockDim.x)
    __builtin_nontemporal_store((c == 0) ? 1.0f : 0.0f, &g[c]);
}

extern "C" void kernel_launch(void* const* d_in, const int* in_sizes, int n_in,
                              void* d_out, int out_size, void* d_ws, size_t ws_size,
                              hipStream_t stream) {
  (void)in_sizes; (void)n_in; (void)out_size; (void)ws_size;
  const float* feat0  = (const float*)d_in[0];
  const float* feat1  = (const float*)d_in[1];
  const float* nfeat0 = (const float*)d_in[2];
  const float* nfeat1 = (const float*)d_in[3];
  const float* conf0  = (const float*)d_in[4];
  const float* conf1  = (const float*)d_in[5];
  const float* nconf0 = (const float*)d_in[6];
  const float* nconf1 = (const float*)d_in[7];
  const float* pos0   = (const float*)d_in[8];
  const float* pos1   = (const float*)d_in[9];
  const int*   idx_q  = (const int*)d_in[10];
  const int*   idx_d  = (const int*)d_in[11];

  float* out = (float*)d_out;
  const size_t bn = BN, sc = SC;
  float* out_scores = out;
  float* out_noise  = out + bn * sc;
  float* out_gt     = out + 2 * bn * sc;
  float* out_mask   = out + 3 * bn * sc;
  float* out_qconf  = out_mask + bn;
  float* out_nqconf = out_qconf + bn;

  _Float16* Ah  = (_Float16*)d_ws;         // vf0  (BN x C)
  _Float16* Bh  = Ah  + (size_t)BN * C;    // dtr
  _Float16* nAh = Bh  + (size_t)BN * C;    // vnf0
  _Float16* nBh = nAh + (size_t)BN * C;    // ndtr
  float* wvp = (float*)(nBh + (size_t)BN * C);
  float* wvp1x = wvp;            float* wvp1y = wvp + BN;
  float* wvp2x = wvp + 2 * BN;   float* wvp2y = wvp + 3 * BN;

  sampler_kernel<<<dim3((BN + 7) / 8, 2), 256, 0, stream>>>(
      feat0, feat1, nfeat0, nfeat1, conf0, conf1, nconf0, nconf1,
      pos0, pos1, idx_q, idx_d,
      out_scores, out_noise, out_mask, out_qconf, out_nqconf,
      Ah, Bh, nAh, nBh, wvp1x, wvp1y, wvp2x, wvp2y);

  gt_kernel<<<BN, 256, 0, stream>>>(out_gt);

  gemm_kernel<<<dim3((BN + 63) / 64, (BN + 63) / 64, 2), 256, 0, stream>>>(
      Ah, Bh, nAh, nBh, wvp1x, wvp1y, wvp2x, wvp2y, out);
}
